// MLA_KV2_Attention_RALATE_32358283608179
// MI455X (gfx1250) — matrix operand order repaired and checked by reading, not compiled, not run on hardware
//
#include <hip/hip_runtime.h>

// ---------------- problem constants (match reference) ----------------
#define B_   2
#define T_   2048
#define H_   16
#define HD_  128
#define DM_  2048      // d_model = n_heads*head_dim
#define DL_  512       // d_latent
#define DC_  256       // d_comp
#define BT_  (B_ * T_)

typedef unsigned short ushort_t;
typedef unsigned long long uptr_t;
typedef __attribute__((ext_vector_type(16))) __bf16        bf16x16;
typedef __attribute__((ext_vector_type(8)))  float         f32x8;
typedef __attribute__((ext_vector_type(4)))  unsigned int  u32x4;
typedef __attribute__((ext_vector_type(4)))  int           i32x4;

union Frag { bf16x16 v; u32x4 q[2]; };

#define AS1 __attribute__((address_space(1)))
#define AS3 __attribute__((address_space(3)))

// ---- gfx1250 async-to-LDS path (probe-guarded; sync LDS staging fallback) ----
#if defined(__has_builtin)
#if __has_builtin(__builtin_amdgcn_global_load_async_to_lds_b128)
#define HAVE_ASYNC_LDS 1
#endif
#endif
#ifndef HAVE_ASYNC_LDS
#define HAVE_ASYNC_LDS 0
#endif

#if defined(__has_builtin)
#if __has_builtin(__builtin_amdgcn_s_wait_asynccnt)
#define WAIT_ASYNC(n) __builtin_amdgcn_s_wait_asynccnt(n)
#endif
#endif
#if !defined(WAIT_ASYNC) && HAVE_ASYNC_LDS
#define WAIT_ASYNC(n) asm volatile("s_wait_asynccnt %0" ::"i"(n) : "memory")
#endif
#ifndef WAIT_ASYNC
#define WAIT_ASYNC(n) ((void)0)
#endif

__device__ __forceinline__ ushort_t f2bf(float f) {
  union { float f; unsigned u; } c; c.f = f;
  unsigned r = c.u + 0x7FFFu + ((c.u >> 16) & 1u);   // round-to-nearest-even
  return (ushort_t)(r >> 16);
}

// ---------------- fp32 -> bf16 convert ----------------
__global__ void cvt_f32_bf16(const float* __restrict__ in,
                             ushort_t* __restrict__ out, int n) {
  int i = blockIdx.x * blockDim.x + threadIdx.x;
  if (i < n) out[i] = f2bf(in[i]);
}

// ---------------- WMMA GEMM:  C[M,N] = A[M,K](bf16) * W[N,K]^T(bf16) + bias ----
// Block = 128 threads (4 waves), block tile 128Mx64N; wave tile 32Mx64N
// (2 A-frags x 4 B-frags -> 8 v_wmma per K-step of 32). The 64x32 W tile is
// shared by all waves: staged into double-buffered LDS with
// global_load_async_to_lds_b128 (ASYNCcnt) and consumed via ds_load_b128.
// M % 128 == 0, N % 64 == 0, K % 32 == 0.
template <bool BF16OUT>
__global__ __launch_bounds__(128) void gemm_wmma(
    const ushort_t* __restrict__ A, const ushort_t* __restrict__ W,
    const float* __restrict__ bias, void* __restrict__ C,
    int M, int N, int K) {
  const int tid  = threadIdx.x;
  const int wave = tid >> 5;
  const int lane = tid & 31;
  const int half = lane >> 4;
  const int l16  = lane & 15;
  const int tileN = blockIdx.x * 64;
  const int tileM = blockIdx.y * 128 + wave * 32;
  (void)M;

  // rows padded to 40 ushorts (80B) to spread LDS banks; 16B aligned chunks
  __shared__ __align__(16) ushort_t Wl[2][64][40];

  f32x8 acc[2][4] = {};
  const ushort_t* arow0 = A + (size_t)(tileM + l16) * K;
  const ushort_t* arow1 = A + (size_t)(tileM + 16 + l16) * K;
  const ushort_t* wtile = W + (size_t)tileN * K;

  // cooperative copy layout: 256 x 16B chunks, 2 per thread
  const int r0 = tid >> 2;            // rows 0..31
  const int r1 = r0 + 32;             // rows 32..63
  const int c0 = (tid & 3) * 8;       // col offset in bf16 elems

  auto stage = [&](int buf, int kk) {
#if HAVE_ASYNC_LDS
    __builtin_amdgcn_global_load_async_to_lds_b128(
        (AS1 i32x4*)(uptr_t)(wtile + (size_t)r0 * K + kk + c0),
        (AS3 i32x4*)&Wl[buf][r0][c0], 0, 0);
    __builtin_amdgcn_global_load_async_to_lds_b128(
        (AS1 i32x4*)(uptr_t)(wtile + (size_t)r1 * K + kk + c0),
        (AS3 i32x4*)&Wl[buf][r1][c0], 0, 0);
#else
    *(u32x4*)&Wl[buf][r0][c0] = *(const u32x4*)(wtile + (size_t)r0 * K + kk + c0);
    *(u32x4*)&Wl[buf][r1][c0] = *(const u32x4*)(wtile + (size_t)r1 * K + kk + c0);
#endif
  };

  stage(0, 0);
  int buf = 0;
  for (int kk = 0; kk < K; kk += 32, buf ^= 1) {
    if (kk + 32 < K) {
      stage(buf ^ 1, kk + 32);          // prefetch next W chunk into other buffer
      WAIT_ASYNC(2);                    // oldest 2 (current buffer) complete
    } else {
      WAIT_ASYNC(0);
    }
    __syncthreads();                    // current buffer visible to all waves

    // A fragments (16x32 bf16): lanes 0-15 K{0..7,16..23}, lanes 16-31 K{8..15,24..31}
    Frag a0, a1;
    a0.q[0] = *(const u32x4*)(arow0 + kk + half * 8);
    a0.q[1] = *(const u32x4*)(arow0 + kk + 16 + half * 8);
    a1.q[0] = *(const u32x4*)(arow1 + kk + half * 8);
    a1.q[1] = *(const u32x4*)(arow1 + kk + 16 + half * 8);
    __builtin_prefetch(arow0 + kk + 512, 0, 1);   // global_prefetch_b8
#pragma unroll
    for (int s = 0; s < 4; ++s) {
      // B fragment (32x16): lane holds w-row (s*16+l16), K = {half*8..+7, 16+half*8..+7}
      Frag b;
      b.q[0] = *(const u32x4*)(&Wl[buf][s * 16 + l16][half * 8]);
      b.q[1] = *(const u32x4*)(&Wl[buf][s * 16 + l16][16 + half * 8]);
      acc[0][s] = __builtin_amdgcn_wmma_f32_16x16x32_bf16(
          false, a0.v, false, b.v, (short)0, acc[0][s], false, false);
      acc[1][s] = __builtin_amdgcn_wmma_f32_16x16x32_bf16(
          false, a1.v, false, b.v, (short)0, acc[1][s], false, false);
    }
    __syncthreads();                    // all reads done before buffer is restaged
  }

#pragma unroll
  for (int g = 0; g < 2; ++g)
#pragma unroll
    for (int s = 0; s < 4; ++s) {
      const int col = tileN + s * 16 + l16;
      const float bv = bias[col];
#pragma unroll
      for (int r = 0; r < 8; ++r) {
        const int row = tileM + g * 16 + r + half * 8;   // C layout: row = r + 8*half
        float v = acc[g][s][r] + bv;
        if (BF16OUT) ((ushort_t*)C)[(size_t)row * N + col] = f2bf(v);
        else         ((float*)C)[(size_t)row * N + col] = v;
      }
    }
}

// ---------------- RoPE + convert to per-head bf16 layout [B,H,T,HD] ----------------
__global__ void rope_cvt(const float* __restrict__ X, ushort_t* __restrict__ Ob) {
  int idx = blockIdx.x * blockDim.x + threadIdx.x;
  const int total = B_ * T_ * H_ * 64;
  if (idx >= total) return;
  int p = idx;
  const int d = p & 63;  p >>= 6;
  const int h = p & 15;  p >>= 4;
  const int t = p % T_;  const int b = p / T_;
  const float ang = (float)t * __powf(10000.0f, -(float)(2 * d) * (1.0f / 128.0f));
  float sn, cs;
  __sincosf(ang, &sn, &cs);
  const float* xr = X + (size_t)(b * T_ + t) * DM_ + h * HD_;
  const float x1 = xr[d], x2 = xr[d + 64];
  ushort_t* orow = Ob + ((size_t)(b * H_ + h) * T_ + t) * HD_;
  orow[d]      = f2bf(x1 * cs - x2 * sn);
  orow[d + 64] = f2bf(x2 * cs + x1 * sn);
}

// ---------------- V transpose: [B*T, H*HD] -> [B,H,HD,T] ----------------
__global__ void transpose_v(const ushort_t* __restrict__ Vin, ushort_t* __restrict__ Vt) {
  int idx = blockIdx.x * blockDim.x + threadIdx.x;
  const int total = B_ * T_ * DM_;
  if (idx >= total) return;
  int p = idx;
  const int c = p & (DM_ - 1); p >>= 11;
  const int t = p % T_;  const int b = p / T_;
  const int h = c >> 7;  const int d = c & 127;
  Vt[(size_t)((b * H_ + h) * HD_ + d) * T_ + t] = Vin[idx];
}

// ---------------- Flash attention (causal), wave32 WMMA ----------------
// "Queries" = RoPE(K), "Keys" = RoPE(Q) (RALATE swap). grid = (T/64, B*H).
// Wave owns 16 query rows; 64-key tiles; S = 16 WMMA, PV = 16 WMMA per tile.
__global__ __launch_bounds__(128) void attn_wmma(
    const ushort_t* __restrict__ Qr,   // [BH][T][HD] query rows (= K tensor)
    const ushort_t* __restrict__ Kr,   // [BH][T][HD] key rows   (= Q tensor)
    const ushort_t* __restrict__ Vt,   // [BH][HD][T]
    ushort_t* __restrict__ O) {        // [B*T][DM] (col = h*HD + d)
  const int bh = blockIdx.y;
  const int b  = bh >> 4;
  const int h  = bh & 15;
  const int i0 = blockIdx.x * 64;
  const int wave = threadIdx.x >> 5;
  const int lane = threadIdx.x & 31;
  const int half = lane >> 4;
  const int l16  = lane & 15;
  const int iw = i0 + wave * 16;

  const ushort_t* Qb = Qr + (size_t)bh * T_ * HD_;
  const ushort_t* Kb = Kr + (size_t)bh * T_ * HD_;
  const ushort_t* Vb = Vt + (size_t)bh * HD_ * T_;

  __shared__ __align__(16) ushort_t P[4][16][72];  // per-wave P tile, padded rows

  // A fragments over head dim: 4 chunks of 32 (persist across the whole j loop)
  Frag af[4];
  const ushort_t* qrow = Qb + (size_t)(iw + l16) * HD_;
#pragma unroll
  for (int f = 0; f < 4; ++f) {
    af[f].q[0] = *(const u32x4*)(qrow + f * 32 + half * 8);
    af[f].q[1] = *(const u32x4*)(qrow + f * 32 + 16 + half * 8);
  }

  f32x8 o[8] = {};
  float mrow[8], lrow[8];
#pragma unroll
  for (int r = 0; r < 8; ++r) { mrow[r] = -1e30f; lrow[r] = 0.f; }
  const float scale = 0.08838834764831845f;   // 1/sqrt(128)

  for (int j0 = 0; j0 <= i0; j0 += 64) {
    // ---- S = Q_tile . K_tile^T : 4 col-subtiles x 4 d-chunks ----
    f32x8 sa[4] = {};
#pragma unroll
    for (int sub = 0; sub < 4; ++sub) {
      const ushort_t* krow = Kb + (size_t)(j0 + sub * 16 + l16) * HD_;
#pragma unroll
      for (int f = 0; f < 4; ++f) {
        Frag bf;
        bf.q[0] = *(const u32x4*)(krow + f * 32 + half * 8);
        bf.q[1] = *(const u32x4*)(krow + f * 32 + 16 + half * 8);
        sa[sub] = __builtin_amdgcn_wmma_f32_16x16x32_bf16(
            false, af[f].v, false, bf.v, (short)0, sa[sub], false, false);
      }
    }
    // ---- scale + causal mask (only diagonal tile needs it) ----
    const bool diag = (j0 == i0);
#pragma unroll
    for (int sub = 0; sub < 4; ++sub) {
      const int col = j0 + sub * 16 + l16;
#pragma unroll
      for (int r = 0; r < 8; ++r) {
        float v = sa[sub][r] * scale;
        if (diag && col > (iw + r + half * 8)) v = -1e30f;
        sa[sub][r] = v;
      }
    }
    // ---- online softmax: row max over 4 subtiles + 16 lanes ----
    float corr[8];
#pragma unroll
    for (int r = 0; r < 8; ++r) {
      float m = fmaxf(fmaxf(sa[0][r], sa[1][r]), fmaxf(sa[2][r], sa[3][r]));
      m = fmaxf(m, __shfl_xor(m, 1, 32));
      m = fmaxf(m, __shfl_xor(m, 2, 32));
      m = fmaxf(m, __shfl_xor(m, 4, 32));
      m = fmaxf(m, __shfl_xor(m, 8, 32));
      const float mn = fmaxf(mrow[r], m);
      corr[r] = __expf(mrow[r] - mn);
      mrow[r] = mn;
    }
    float rs[8] = {0, 0, 0, 0, 0, 0, 0, 0};
#pragma unroll
    for (int sub = 0; sub < 4; ++sub)
#pragma unroll
      for (int r = 0; r < 8; ++r) {
        const float e = __expf(sa[sub][r] - mrow[r]);
        rs[r] += e;
        P[wave][r + 8 * half][sub * 16 + l16] = f2bf(e);   // C->A relayout via LDS
      }
#pragma unroll
    for (int r = 0; r < 8; ++r) {
      float t = rs[r];
      t += __shfl_xor(t, 1, 32);
      t += __shfl_xor(t, 2, 32);
      t += __shfl_xor(t, 4, 32);
      t += __shfl_xor(t, 8, 32);
      lrow[r] = lrow[r] * corr[r] + t;
    }
#pragma unroll
    for (int d = 0; d < 8; ++d)
#pragma unroll
      for (int r = 0; r < 8; ++r) o[d][r] *= corr[r];
    // ---- O += P . V : 2 key-chunks x 8 head-dim subtiles ----
#pragma unroll
    for (int ks = 0; ks < 64; ks += 32) {
      Frag pa;
      pa.q[0] = *(const u32x4*)(&P[wave][l16][ks + half * 8]);
      pa.q[1] = *(const u32x4*)(&P[wave][l16][ks + 16 + half * 8]);
#pragma unroll
      for (int d = 0; d < 8; ++d) {
        const ushort_t* vrow = Vb + (size_t)(d * 16 + l16) * T_ + j0 + ks + half * 8;
        Frag vb;
        vb.q[0] = *(const u32x4*)(vrow);
        vb.q[1] = *(const u32x4*)(vrow + 16);
        o[d] = __builtin_amdgcn_wmma_f32_16x16x32_bf16(
            false, pa.v, false, vb.v, (short)0, o[d], false, false);
      }
    }
  }
  // ---- finalize: divide by l, write [B,T,H*HD] bf16 ----
#pragma unroll
  for (int d = 0; d < 8; ++d)
#pragma unroll
    for (int r = 0; r < 8; ++r) {
      const int trow = iw + r + 8 * half;
      O[(size_t)(b * T_ + trow) * DM_ + h * HD_ + d * 16 + l16] =
          f2bf(o[d][r] * (1.0f / lrow[r]));
    }
}

// ---------------- host orchestration ----------------
extern "C" void kernel_launch(void* const* d_in, const int* in_sizes, int n_in,
                              void* d_out, int out_size, void* d_ws, size_t ws_size,
                              hipStream_t stream) {
  (void)in_sizes; (void)n_in; (void)out_size; (void)ws_size;
  const float* x        = (const float*)d_in[0];
  const float* Wq_w     = (const float*)d_in[1];
  const float* Wq_b     = (const float*)d_in[2];
  const float* tok_w    = (const float*)d_in[3];
  const float* tok_b    = (const float*)d_in[4];
  const float* kc_w     = (const float*)d_in[5];
  const float* kc_b     = (const float*)d_in[6];
  const float* kd_w     = (const float*)d_in[7];
  const float* kd_b     = (const float*)d_in[8];
  const float* fromk_w  = (const float*)d_in[9];
  const float* fromk_b  = (const float*)d_in[10];
  const float* tov_w    = (const float*)d_in[11];
  const float* tov_b    = (const float*)d_in[12];
  const float* vc_w     = (const float*)d_in[13];
  const float* vc_b     = (const float*)d_in[14];
  const float* vd_w     = (const float*)d_in[15];
  const float* vd_b     = (const float*)d_in[16];
  const float* fromv_w  = (const float*)d_in[17];
  const float* fromv_b  = (const float*)d_in[18];
  const float* out_w    = (const float*)d_in[19];
  const float* out_b    = (const float*)d_in[20];
  float* out = (float*)d_out;

  char* ws = (char*)d_ws;
  size_t off = 0;
  auto alloc = [&](size_t bytes) -> char* {
    char* p = ws + off;
    off += (bytes + 255) & ~(size_t)255;
    return p;
  };
  ushort_t* xb      = (ushort_t*)alloc((size_t)BT_ * DM_ * 2);
  ushort_t* wq16    = (ushort_t*)alloc((size_t)DM_ * DM_ * 2);
  ushort_t* tok16   = (ushort_t*)alloc((size_t)DL_ * DM_ * 2);
  ushort_t* kc16    = (ushort_t*)alloc((size_t)DC_ * DL_ * 2);
  ushort_t* kd16    = (ushort_t*)alloc((size_t)DL_ * DC_ * 2);
  ushort_t* fromk16 = (ushort_t*)alloc((size_t)DM_ * DL_ * 2);
  ushort_t* tov16   = (ushort_t*)alloc((size_t)DL_ * DM_ * 2);
  ushort_t* vc16    = (ushort_t*)alloc((size_t)DC_ * DL_ * 2);
  ushort_t* vd16    = (ushort_t*)alloc((size_t)DL_ * DC_ * 2);
  ushort_t* fromv16 = (ushort_t*)alloc((size_t)DM_ * DL_ * 2);
  ushort_t* outw16  = (ushort_t*)alloc((size_t)DM_ * DM_ * 2);
  float*    q_f32   = (float*)alloc((size_t)BT_ * DM_ * 4);
  float*    k_f32   = (float*)alloc((size_t)BT_ * DM_ * 4);
  ushort_t* t1      = (ushort_t*)alloc((size_t)BT_ * DL_ * 2);
  ushort_t* t2      = (ushort_t*)alloc((size_t)BT_ * DC_ * 2);
  ushort_t* lat     = (ushort_t*)alloc((size_t)BT_ * DL_ * 2);
  ushort_t* v_bf    = (ushort_t*)alloc((size_t)BT_ * DM_ * 2);
  ushort_t* q_bf    = (ushort_t*)alloc((size_t)BT_ * DM_ * 2);
  ushort_t* k_bf    = (ushort_t*)alloc((size_t)BT_ * DM_ * 2);
  ushort_t* vt      = (ushort_t*)alloc((size_t)BT_ * DM_ * 2);
  ushort_t* attn_bf = (ushort_t*)alloc((size_t)BT_ * DM_ * 2);

  auto cvt = [&](const float* in, ushort_t* o, int n) {
    cvt_f32_bf16<<<dim3((n + 255) / 256), dim3(256), 0, stream>>>(in, o, n);
  };
  cvt(x,       xb,      BT_ * DM_);
  cvt(Wq_w,    wq16,    DM_ * DM_);
  cvt(tok_w,   tok16,   DL_ * DM_);
  cvt(kc_w,    kc16,    DC_ * DL_);
  cvt(kd_w,    kd16,    DL_ * DC_);
  cvt(fromk_w, fromk16, DM_ * DL_);
  cvt(tov_w,   tov16,   DL_ * DM_);
  cvt(vc_w,    vc16,    DC_ * DL_);
  cvt(vd_w,    vd16,    DL_ * DC_);
  cvt(fromv_w, fromv16, DM_ * DL_);
  cvt(out_w,   outw16,  DM_ * DM_);

  auto gemm_f32 = [&](const ushort_t* A, const ushort_t* W, const float* b,
                      float* C, int M, int N, int K) {
    gemm_wmma<false><<<dim3(N / 64, M / 128), dim3(128), 0, stream>>>(A, W, b, C, M, N, K);
  };
  auto gemm_bf = [&](const ushort_t* A, const ushort_t* W, const float* b,
                     ushort_t* C, int M, int N, int K) {
    gemm_wmma<true><<<dim3(N / 64, M / 128), dim3(128), 0, stream>>>(A, W, b, C, M, N, K);
  };

  // Q projection
  gemm_f32(xb, wq16, Wq_b, q_f32, BT_, DM_, DM_);
  // K latent path: tok -> kc -> kd -> fromk
  gemm_bf(xb,  tok16, tok_b, t1,  BT_, DL_, DM_);
  gemm_bf(t1,  kc16,  kc_b,  t2,  BT_, DC_, DL_);
  gemm_bf(t2,  kd16,  kd_b,  lat, BT_, DL_, DC_);
  gemm_f32(lat, fromk16, fromk_b, k_f32, BT_, DM_, DL_);
  // V latent path: tov -> vc -> vd -> fromv
  gemm_bf(xb,  tov16, tov_b, t1,  BT_, DL_, DM_);
  gemm_bf(t1,  vc16,  vc_b,  t2,  BT_, DC_, DL_);
  gemm_bf(t2,  vd16,  vd_b,  lat, BT_, DL_, DC_);
  gemm_bf(lat, fromv16, fromv_b, v_bf, BT_, DM_, DL_);

  // RoPE (+ layout [B,H,T,HD] bf16)
  {
    const int n = B_ * T_ * H_ * 64;
    rope_cvt<<<dim3((n + 255) / 256), dim3(256), 0, stream>>>(q_f32, q_bf);
    rope_cvt<<<dim3((n + 255) / 256), dim3(256), 0, stream>>>(k_f32, k_bf);
  }
  // V -> [B,H,HD,T]
  {
    const int n = B_ * T_ * DM_;
    transpose_v<<<dim3((n + 255) / 256), dim3(256), 0, stream>>>(v_bf, vt);
  }
  // Causal SDPA with RALATE swap: queries = K, keys = Q
  attn_wmma<<<dim3(T_ / 64, B_ * H_), dim3(128), 0, stream>>>(k_bf, q_bf, vt, attn_bf);

  // Output projection (fp32 result)
  gemm_f32(attn_bf, outw16, out_b, out, BT_, DM_, DM_);
}
